// Model_20555713478965
// MI455X (gfx1250) — compile-verified, matrix operand order checked by reading, not run on hardware
//
#include <hip/hip_runtime.h>
#include <hip/hip_bf16.h>

typedef _Float16 h16;
typedef __attribute__((ext_vector_type(16))) _Float16 v16h;
typedef __attribute__((ext_vector_type(8)))  _Float16 v8h;
typedef __attribute__((ext_vector_type(8)))  float    v8f;

// Problem constants (match reference)
#define N_NEIGH 64
#define D_FEAT  64
#define H_OUT   128
#define DECAY_W 2.0f
#define EPS_N   1e-12f

// Branchless fast tanh entirely on the TRANS pipe:
//   tanh(x) = 1 - 2*rcp(exp(2x)+1)
// v_exp_f32 + v_rcp_f32, no div expansion, no exec-mask branches.
// Saturates correctly: exp->inf => 1-0, exp->0 => 1-2 = -1.
__device__ __forceinline__ float fast_tanhf(float x) {
    float e = __expf(2.0f * x);                      // v_exp_f32
    float r = __builtin_amdgcn_rcpf(e + 1.0f);       // v_rcp_f32
    return 1.0f - 2.0f * r;
}

// LDS layout. f16 matrices use row stride 72 halves (144 B): keeps 16 B
// alignment for v8h loads and makes the 16-lane row stride cover all 64 LDS
// banks (36*r mod 64 distinct for r=0..15) -> conflict-free ds_load_b128.
struct __align__(16) Smem {
    alignas(16) h16 neighH[64 * 72];   // neigh_f in f16 (WMMA A operand)
    alignas(16) h16 w2t[64 * 72];      // W2^T in f16  (WMMA B operand, col-major per e)
    float neighF[64 * 65];             // neigh_f f32 (stride 65: conflict-free col reads)
    float t2vW[64], t2vB[64], nodeWv[64], nodeBv[64], attV[64];
    float qv[64];                      // node_f @ W1
    float nodeF[64];
    float attS[64];
    float wn[64];
    float dtN[64], degN[64], ccN[64], maskF[64];
    float combined[256];
    float nIn;
    float scaleNode;
};

__global__ void __launch_bounds__(128)
tgn_fused_kernel(const float* __restrict__ dt_node,
                 const float* __restrict__ deg_node,
                 const float* __restrict__ cc_node,
                 const float* __restrict__ dt_neigh,
                 const float* __restrict__ deg_neigh,
                 const float* __restrict__ cc_neigh,
                 const int*   __restrict__ neigh_mask,
                 const float* __restrict__ feature_hist,
                 const float* __restrict__ t2v_w,
                 const float* __restrict__ t2v_b,
                 const float* __restrict__ node_w,
                 const float* __restrict__ node_b,
                 const float* __restrict__ att_W1,
                 const float* __restrict__ att_W2,
                 const float* __restrict__ att_v,
                 const float* __restrict__ weight,
                 float* __restrict__ out)
{
    __shared__ Smem s;
    const int b    = blockIdx.x;
    const int tid  = threadIdx.x;
    const int lane = tid & 31;
    const int wave = tid >> 5;

    // ---- stage small vectors and per-neighbor scalars ----
    if (tid < 64) {
        s.t2vW[tid]   = t2v_w[tid];
        s.t2vB[tid]   = t2v_b[tid];
        s.nodeWv[tid] = node_w[tid];
        s.nodeBv[tid] = node_b[tid];
        s.attV[tid]   = att_v[tid];
        s.dtN[tid]    = dt_neigh[b * N_NEIGH + tid];
        s.degN[tid]   = deg_neigh[b * N_NEIGH + tid];
        s.ccN[tid]    = cc_neigh[b * N_NEIGH + tid];
        s.maskF[tid]  = (neigh_mask[b * N_NEIGH + tid] != 0) ? 1.0f : 0.0f;
    }
    __syncthreads();

    // ---- node feature (raw) ----
    if (tid < 64) {
        float dt = fabsf(dt_node[b]);
        float dc = deg_node[b] + cc_node[b];
        float v  = dt * s.t2vW[tid] + s.t2vB[tid];
        float t2 = (tid == 0) ? v : __cosf(v);       // v_cos_f32 (TRANS)
        s.nodeF[tid] = t2 + dc * s.nodeWv[tid] + 2.0f * s.nodeBv[tid];
    }
    __syncthreads();
    if (tid == 0) {
        float ss = 0.0f;
        for (int d = 0; d < 64; ++d) ss += s.nodeF[d] * s.nodeF[d];
        s.scaleNode = 1.0f / fmaxf(sqrtf(ss), EPS_N);
        float ms = 0.0f;
        for (int n = 0; n < 64; ++n) ms += s.maskF[n];
        s.nIn = fmaxf(ms, 1.0f);
    }
    __syncthreads();
    if (tid < 64) s.nodeF[tid] *= s.scaleNode;
    __syncthreads();

    // ---- q = node_f @ W1 (folded into WMMA C operand later) ----
    if (tid < 64) {
        float acc = 0.0f;
        for (int d = 0; d < 64; ++d) acc += s.nodeF[d] * att_W1[d * 64 + tid];
        s.qv[tid] = acc;
    }

    // ---- neighbor features (raw) ----
    for (int i = tid; i < 64 * 64; i += 128) {
        int n = i >> 6, d = i & 63;
        float dt = fabsf(s.dtN[n]);
        float dc = s.degN[n] + s.ccN[n];
        float v  = dt * s.t2vW[d] + s.t2vB[d];
        float t2 = (d == 0) ? v : __cosf(v);         // v_cos_f32 (TRANS)
        s.neighF[n * 65 + d] = t2 + dc * s.nodeWv[d] + 2.0f * s.nodeBv[d];
    }
    // ---- W2^T into f16 LDS (B operand: per output column e, K contiguous) ----
    for (int i = tid; i < 64 * 64; i += 128) {
        int e = i >> 6, k = i & 63;
        s.w2t[e * 72 + k] = (h16)att_W2[k * 64 + e];
    }
    __syncthreads();

    // ---- per-row L2 normalize, emit f16 copy for WMMA ----
    if (tid < 64) {
        const int n = tid;
        float ss = 0.0f;
        for (int d = 0; d < 64; ++d) { float f = s.neighF[n * 65 + d]; ss += f * f; }
        float sc = 1.0f / fmaxf(sqrtf(ss), EPS_N);
        for (int d = 0; d < 64; ++d) {
            float f = s.neighF[n * 65 + d] * sc;
            s.neighF[n * 65 + d] = f;
            s.neighH[n * 72 + d] = (h16)f;
        }
    }
    __syncthreads();

    // ---- WMMA: h_pre[16x64 stripe] = q (C) + neigh_f (A) @ W2 (B) ----
    // wave w owns rows [w*16, w*16+16). 4 N-tiles x 2 K-steps = 8 wmma/wave.
    {
        const int mrow  = wave * 16;
        const int r16   = lane & 15;   // A: row-in-tile; B/C/D: col-in-tile
        const int khalf = lane >> 4;   // which K half-chunk this lane holds
        float attAcc[8] = {0.f, 0.f, 0.f, 0.f, 0.f, 0.f, 0.f, 0.f};

        #pragma unroll
        for (int tn = 0; tn < 4; ++tn) {
            v8f c;
            const float qcol = s.qv[tn * 16 + r16];
            #pragma unroll
            for (int r = 0; r < 8; ++r) c[r] = qcol;

            #pragma unroll
            for (int ks = 0; ks < 2; ++ks) {
                // A fragment: lane holds row (mrow+r16), K = ks*32 + khalf*8 + [0..7]
                // and K = ks*32 + 16 + khalf*8 + [0..7]  (ISA 16-bit A layout)
                const h16* aP = &s.neighH[(mrow + r16) * 72 + ks * 32 + khalf * 8];
                v8h alo = *(const v8h*)aP;
                v8h ahi = *(const v8h*)(aP + 16);
                v16h a = __builtin_shufflevector(alo, ahi,
                         0, 1, 2, 3, 4, 5, 6, 7, 8, 9, 10, 11, 12, 13, 14, 15);

                // B fragment: lane holds col (tn*16+r16), same K pattern (from W2^T)
                const h16* bP = &s.w2t[(tn * 16 + r16) * 72 + ks * 32 + khalf * 8];
                v8h blo = *(const v8h*)bP;
                v8h bhi = *(const v8h*)(bP + 16);
                v16h bm = __builtin_shufflevector(blo, bhi,
                          0, 1, 2, 3, 4, 5, 6, 7, 8, 9, 10, 11, 12, 13, 14, 15);

                c = __builtin_amdgcn_wmma_f32_16x16x32_f16(
                        /*neg_a=*/false, a, /*neg_b=*/false, bm,
                        /*c_mod=*/(short)0, c, /*reuse_a=*/false, /*reuse_b=*/false);
            }

            // att contribution: tanh(h) * v[col], reduce over the 16 cols of this tile.
            const float vcol = s.attV[tn * 16 + r16];
            #pragma unroll
            for (int r = 0; r < 8; ++r) {
                float t = fast_tanhf(c[r]) * vcol;   // exp+rcp, branchless
                t += __shfl_xor(t, 1);
                t += __shfl_xor(t, 2);
                t += __shfl_xor(t, 4);
                t += __shfl_xor(t, 8);   // stays within 16-lane half (row identity)
                attAcc[r] += t;
            }
        }
        // VGPR r: lanes 0-15 -> row mrow+r, lanes 16-31 -> row mrow+8+r
        if (r16 == 0) {
            #pragma unroll
            for (int r = 0; r < 8; ++r) s.attS[mrow + khalf * 8 + r] = attAcc[r];
        }
    }
    __syncthreads();

    // ---- decay * leaky_relu * att, masked, /n_in ----
    if (tid < 64) {
        float ts = 1.0f / (1.0f + DECAY_W * s.dtN[tid]);
        float lr = (ts > 0.0f) ? ts : 0.01f * ts;
        float score = lr * s.attS[tid];
        s.wn[tid] = s.maskF[tid] * score / s.nIn;
    }
    __syncthreads();

    // ---- neigh_agg[d] = sum_n wn[n]*neigh_f[n,d]; build combined[256] ----
    if (tid < 64) {
        float acc = 0.0f;
        for (int n = 0; n < 64; ++n) acc += s.wn[n] * s.neighF[n * 65 + tid];
        s.combined[tid]      = s.nodeF[tid];
        s.combined[64 + tid] = acc;
    }
    s.combined[128 + tid] = feature_hist[b * 128 + tid];
    __syncthreads();

    // ---- out[h] = relu(combined . weight[h,:]) ----
    {
        float acc = 0.0f;
        const float4* wrow = (const float4*)(weight + tid * 256);
        #pragma unroll 4
        for (int k4 = 0; k4 < 64; ++k4) {
            float4 wv = wrow[k4];
            acc += wv.x * s.combined[k4 * 4 + 0];
            acc += wv.y * s.combined[k4 * 4 + 1];
            acc += wv.z * s.combined[k4 * 4 + 2];
            acc += wv.w * s.combined[k4 * 4 + 3];
        }
        out[b * H_OUT + tid] = fmaxf(acc, 0.0f);
    }
}

extern "C" void kernel_launch(void* const* d_in, const int* in_sizes, int n_in,
                              void* d_out, int out_size, void* d_ws, size_t ws_size,
                              hipStream_t stream) {
    const float* dt_node      = (const float*)d_in[0];
    const float* deg_node     = (const float*)d_in[1];
    const float* cc_node      = (const float*)d_in[2];
    const float* dt_neigh     = (const float*)d_in[3];
    const float* deg_neigh    = (const float*)d_in[4];
    const float* cc_neigh     = (const float*)d_in[5];
    const int*   neigh_mask   = (const int*)d_in[6];
    const float* feature_hist = (const float*)d_in[7];
    const float* t2v_w        = (const float*)d_in[8];
    const float* t2v_b        = (const float*)d_in[9];
    const float* node_w       = (const float*)d_in[10];
    const float* node_b       = (const float*)d_in[11];
    const float* att_W1       = (const float*)d_in[12];
    const float* att_W2       = (const float*)d_in[13];
    const float* att_v        = (const float*)d_in[14];
    const float* weight       = (const float*)d_in[15];
    float* out                = (float*)d_out;

    const int B = in_sizes[0];   // 16384
    tgn_fused_kernel<<<dim3(B), dim3(128), 0, stream>>>(
        dt_node, deg_node, cc_node, dt_neigh, deg_neigh, cc_neigh,
        neigh_mask, feature_hist, t2v_w, t2v_b, node_w, node_b,
        att_W1, att_W2, att_v, weight, out);
}
